// CrossAttention_58969900974442
// MI455X (gfx1250) — matrix operand order repaired and checked by reading, not compiled, not run on hardware
//
#include <hip/hip_runtime.h>
#include <hip/hip_bf16.h>
#include <stdint.h>

typedef __attribute__((ext_vector_type(16))) __bf16 v16bf;
typedef __attribute__((ext_vector_type(8)))  float  v8f;

union FragBf {
    v16bf v;
    __bf16 e[16];
    uint4 q[2];
};

// ---------------------------------------------------------------------------
// C = A[MxK] * W[KxN] + bias, fp32 in/out, bf16 WMMA compute.
// Block tile 128x128, 8 waves, each wave 32(M) x 64(N) = 8 wmma tiles.
// Double-buffered LDS, register-staged global loads, 1 barrier / K-step.
// K must be a multiple of 32, N a multiple of 128.
// ---------------------------------------------------------------------------
__global__ __launch_bounds__(256) void gemm_bias_kernel(
    const float* __restrict__ A, const float* __restrict__ W,
    const float* __restrict__ bias, float* __restrict__ C,
    int M, int N, int K)
{
    __shared__ __bf16 As[2][128][40];   // [m][k]   row stride 80B (16B multiple)
    __shared__ __bf16 Bs[2][128][40];   // [n][k]   transposed, K-contiguous frags

    const int t     = threadIdx.x;
    const int lane  = t & 31;
    const int wave  = t >> 5;
    const int wm    = wave & 3;      // 4 waves along M
    const int wn    = wave >> 2;     // 2 waves along N
    const int mbase = blockIdx.x * 128;
    const int nbase = blockIdx.y * 128;
    const int lhalf = (lane < 16) ? 0 : 1;
    const int l16   = lane & 15;
    const int nsteps = K >> 5;

    v8f acc[2][4] = {};
    float4 ra[4], rb[4];             // register-staged next tile

    // batched, branch-free global loads for one 128x32 A slab + 32x128 W slab
    auto load_regs = [&](int step) {
#pragma unroll
        for (int i = 0; i < 4; ++i) {
            int idx  = i * 256 + t;              // 1024 float4 slots
            int row  = idx >> 3;                 // 0..127
            int c4   = idx & 7;                  // 0..7 (x4 cols)
            int grow = mbase + row;
            int crow = (grow < M) ? grow : (M - 1);   // clamp, no exec branch
            float4 f = *(const float4*)(A + (size_t)crow * K + (step << 5) + c4 * 4);
            if (grow >= M) f = make_float4(0.f, 0.f, 0.f, 0.f); // v_cndmask
            ra[i] = f;
        }
#pragma unroll
        for (int i = 0; i < 4; ++i) {
            int idx  = i * 256 + t;
            int krow = idx >> 5;                 // 0..31
            int c4   = idx & 31;                 // 0..31 (x4 cols)
            rb[i] = *(const float4*)(W + (size_t)((step << 5) + krow) * N + nbase + c4 * 4);
        }
    };

    auto store_lds = [&](int buf) {
#pragma unroll
        for (int i = 0; i < 4; ++i) {
            int idx = i * 256 + t;
            int row = idx >> 3, c4 = idx & 7;
            __bf16* dst = &As[buf][row][c4 * 4];
            dst[0] = (__bf16)ra[i].x; dst[1] = (__bf16)ra[i].y;
            dst[2] = (__bf16)ra[i].z; dst[3] = (__bf16)ra[i].w;
        }
#pragma unroll
        for (int i = 0; i < 4; ++i) {
            int idx  = i * 256 + t;
            int krow = idx >> 5, c4 = idx & 31;
            Bs[buf][c4 * 4 + 0][krow] = (__bf16)rb[i].x;
            Bs[buf][c4 * 4 + 1][krow] = (__bf16)rb[i].y;
            Bs[buf][c4 * 4 + 2][krow] = (__bf16)rb[i].z;
            Bs[buf][c4 * 4 + 3][krow] = (__bf16)rb[i].w;
        }
    };

    // pipeline prologue
    load_regs(0);
    store_lds(0);
    if (nsteps > 1) load_regs(1);
    __syncthreads();

    for (int step = 0; step < nsteps; ++step) {
        const int cur = step & 1;

        // write the tile for step+1 into the other buffer (read of that buffer
        // finished before the barrier that ended step-1)
        if (step + 1 < nsteps) store_lds(cur ^ 1);

        // fragments from current buffer
        FragBf a[2], b[4];
#pragma unroll
        for (int i = 0; i < 2; ++i) {
            const __bf16* src = &As[cur][wm * 32 + i * 16 + l16][lhalf * 8];
            a[i].q[0] = *(const uint4*)(src);
            a[i].q[1] = *(const uint4*)(src + 16);
        }
#pragma unroll
        for (int j = 0; j < 4; ++j) {
            const __bf16* src = &Bs[cur][wn * 64 + j * 16 + l16][lhalf * 8];
            b[j].q[0] = *(const uint4*)(src);
            b[j].q[1] = *(const uint4*)(src + 16);
        }
#pragma unroll
        for (int i = 0; i < 2; ++i)
#pragma unroll
            for (int j = 0; j < 4; ++j)
                acc[i][j] = __builtin_amdgcn_wmma_f32_16x16x32_bf16(
                    false, a[i].v, false, b[j].v, (short)0, acc[i][j], false, false);

        // issue HBM loads for step+2 under the WMMAs
        if (step + 2 < nsteps) load_regs(step + 2);

        __syncthreads();
    }

    // epilogue: bias + store
#pragma unroll
    for (int j = 0; j < 4; ++j) {
        int col  = nbase + wn * 64 + j * 16 + l16;
        float bb = bias[col];
#pragma unroll
        for (int i = 0; i < 2; ++i) {
            int rb2 = mbase + wm * 32 + i * 16 + lhalf * 8;
#pragma unroll
            for (int v = 0; v < 8; ++v) {
                int row = rb2 + v;
                if (row < M) C[(size_t)row * N + col] = acc[i][j][v] + bb;
            }
        }
    }
}

// ---------------------------------------------------------------------------
// Fused attention: per (b,h) block, 128 q-rows per block (16 per wave).
// scores = Q Kt / 8 -> softmax over 77 keys (padded to 80) -> P V.
// ---------------------------------------------------------------------------
__global__ __launch_bounds__(256) void attn_kernel(
    const float* __restrict__ Q, const float* __restrict__ Kb,
    const float* __restrict__ Vb, float* __restrict__ O)
{
    __shared__ __bf16 Kl[80][72];        // [key][d]  stride 144B
    __shared__ __bf16 Vt[64][104];       // [d][key]  keys padded 77..95 = 0
    __shared__ __bf16 At[8][16][104];    // per-wave probs [row][key], padded

    const int bh    = blockIdx.x;        // b*16 + h
    const int b     = bh >> 4;
    const int h     = bh & 15;
    const int t     = threadIdx.x;
    const int lane  = t & 31;
    const int wave  = t >> 5;
    const int lhalf = (lane < 16) ? 0 : 1;
    const int l16   = lane & 15;

    // stage K (padded to 80 keys) and V transposed (padded to 96 keys)
    for (int idx = t; idx < 80 * 64; idx += 256) {
        int key = idx >> 6, d = idx & 63;
        float v = (key < 77) ? Kb[(size_t)(b * 77 + key) * 1024 + h * 64 + d] : 0.f;
        Kl[key][d] = (__bf16)v;
    }
    for (int idx = t; idx < 96 * 64; idx += 256) {
        int key = idx >> 6, d = idx & 63;
        float v = (key < 77) ? Vb[(size_t)(b * 77 + key) * 1024 + h * 64 + d] : 0.f;
        Vt[d][key] = (__bf16)v;
    }
    __syncthreads();

    const int qrow0 = blockIdx.y * 128 + wave * 16;

    // Q fragments straight from global (vectorized fp32 loads -> bf16 regs)
    FragBf a[2];
    {
        const float* qp = Q + (size_t)(b * 4096 + qrow0 + l16) * 1024 + h * 64;
#pragma unroll
        for (int ds = 0; ds < 2; ++ds) {
            int base = ds * 32 + lhalf * 8;
            float4 f0 = *(const float4*)(qp + base);
            float4 f1 = *(const float4*)(qp + base + 4);
            float4 f2 = *(const float4*)(qp + base + 16);
            float4 f3 = *(const float4*)(qp + base + 20);
            a[ds].e[0]  = (__bf16)f0.x; a[ds].e[1]  = (__bf16)f0.y;
            a[ds].e[2]  = (__bf16)f0.z; a[ds].e[3]  = (__bf16)f0.w;
            a[ds].e[4]  = (__bf16)f1.x; a[ds].e[5]  = (__bf16)f1.y;
            a[ds].e[6]  = (__bf16)f1.z; a[ds].e[7]  = (__bf16)f1.w;
            a[ds].e[8]  = (__bf16)f2.x; a[ds].e[9]  = (__bf16)f2.y;
            a[ds].e[10] = (__bf16)f2.z; a[ds].e[11] = (__bf16)f2.w;
            a[ds].e[12] = (__bf16)f3.x; a[ds].e[13] = (__bf16)f3.y;
            a[ds].e[14] = (__bf16)f3.z; a[ds].e[15] = (__bf16)f3.w;
        }
    }

    // scores (5 key tiles x 2 d-steps = 10 WMMAs)
    v8f s[5] = {};
#pragma unroll
    for (int kt = 0; kt < 5; ++kt) {
#pragma unroll
        for (int ds = 0; ds < 2; ++ds) {
            FragBf bk;
            const __bf16* src = &Kl[kt * 16 + l16][ds * 32 + lhalf * 8];
            bk.q[0] = *(const uint4*)(src);
            bk.q[1] = *(const uint4*)(src + 16);
            s[kt] = __builtin_amdgcn_wmma_f32_16x16x32_bf16(
                false, a[ds].v, false, bk.v, (short)0, s[kt], false, false);
        }
    }

    // row-wise softmax; row group = 16 lanes (xor masks <= 8 stay in group)
#pragma unroll
    for (int v = 0; v < 8; ++v) {
        float val[5];
        float m = -3.0e38f;
#pragma unroll
        for (int kt = 0; kt < 5; ++kt) {
            int col = kt * 16 + l16;
            val[kt] = (col < 77) ? s[kt][v] * 0.125f : -3.0e38f;
            m = fmaxf(m, val[kt]);
        }
#pragma unroll
        for (int off = 8; off >= 1; off >>= 1) m = fmaxf(m, __shfl_xor(m, off, 32));
        float sum = 0.f, ex[5];
#pragma unroll
        for (int kt = 0; kt < 5; ++kt) {
            ex[kt] = (kt * 16 + l16 < 77) ? __expf(val[kt] - m) : 0.f;
            sum += ex[kt];
        }
#pragma unroll
        for (int off = 8; off >= 1; off >>= 1) sum += __shfl_xor(sum, off, 32);
        float r  = 1.f / sum;
        int  row = v + lhalf * 8;
#pragma unroll
        for (int kt = 0; kt < 5; ++kt) At[wave][row][kt * 16 + l16] = (__bf16)(ex[kt] * r);
        At[wave][row][80 + l16] = (__bf16)0.f;   // zero pad keys 80..95
    }
    __syncthreads();

    // out = P[16x96] * V[96x64]  (3 k-steps x 4 d tiles = 12 WMMAs)
    v8f o[4] = {};
#pragma unroll
    for (int ks = 0; ks < 3; ++ks) {
        FragBf ap;
        const __bf16* asrc = &At[wave][l16][ks * 32 + lhalf * 8];
        ap.q[0] = *(const uint4*)(asrc);
        ap.q[1] = *(const uint4*)(asrc + 16);
#pragma unroll
        for (int j = 0; j < 4; ++j) {
            FragBf bv;
            const __bf16* bsrc = &Vt[j * 16 + l16][ks * 32 + lhalf * 8];
            bv.q[0] = *(const uint4*)(bsrc);
            bv.q[1] = *(const uint4*)(bsrc + 16);
            o[j] = __builtin_amdgcn_wmma_f32_16x16x32_bf16(
                false, ap.v, false, bv.v, (short)0, o[j], false, false);
        }
    }

    // store attention output in [B, Sq, H*D] layout for the final GEMM
#pragma unroll
    for (int j = 0; j < 4; ++j) {
        int col = h * 64 + j * 16 + l16;
        int rb  = qrow0 + lhalf * 8;
#pragma unroll
        for (int v = 0; v < 8; ++v)
            O[(size_t)(b * 4096 + rb + v) * 1024 + col] = o[j][v];
    }
}

// ---------------------------------------------------------------------------
extern "C" void kernel_launch(void* const* d_in, const int* in_sizes, int n_in,
                              void* d_out, int out_size, void* d_ws, size_t ws_size,
                              hipStream_t stream)
{
    const float* x  = (const float*)d_in[0];
    const float* y  = (const float*)d_in[1];
    const float* wq = (const float*)d_in[2];
    const float* bq = (const float*)d_in[3];
    const float* wk = (const float*)d_in[4];
    const float* bk = (const float*)d_in[5];
    const float* wv = (const float*)d_in[6];
    const float* bv = (const float*)d_in[7];
    const float* wo = (const float*)d_in[8];
    const float* bo = (const float*)d_in[9];
    float* out = (float*)d_out;

    const size_t MQ  = 32768;          // 8 * 4096
    const size_t MKV = 616;            // 8 * 77
    float* qbuf = (float*)d_ws;
    float* kbuf = qbuf + MQ * 1024;
    float* vbuf = kbuf + MKV * 1024;
    float* abuf = vbuf + MKV * 1024;   // attention output [B,Sq,1024]

    dim3 blk(256);
    // projections
    gemm_bias_kernel<<<dim3(256, 8), blk, 0, stream>>>(x, wq, bq, qbuf, 32768, 1024, 1024);
    gemm_bias_kernel<<<dim3(5,   8), blk, 0, stream>>>(y, wk, bk, kbuf, 616,   1024, 768);
    gemm_bias_kernel<<<dim3(5,   8), blk, 0, stream>>>(y, wv, bv, vbuf, 616,   1024, 768);
    // fused attention: grid = (B*H, Sq/128)
    attn_kernel<<<dim3(128, 32), blk, 0, stream>>>(qbuf, kbuf, vbuf, abuf);
    // output projection
    gemm_bias_kernel<<<dim3(256, 8), blk, 0, stream>>>(abuf, wo, bo, out, 32768, 1024, 1024);
}
